// PointNetSetAbstractionMsg_60017872994424
// MI455X (gfx1250) — matrix operand order repaired and checked by reading, not compiled, not run on hardware
//
#include <hip/hip_runtime.h>
#include <stdint.h>

#define BB 4
#define NN 4096
#define SS 1024

typedef __attribute__((ext_vector_type(16))) _Float16 v16h;
typedef __attribute__((ext_vector_type(8)))  _Float16 v8h;
typedef __attribute__((ext_vector_type(8)))  float    v8f;

// ---------------- FPS: one block per batch, 1024 threads (32 waves) ----------------
__global__ __launch_bounds__(1024) void fps_kernel(const float* __restrict__ xyz,
                                                   float* __restrict__ query,
                                                   float* __restrict__ outq) {
  int b = blockIdx.x, tid = threadIdx.x;
  const float* xb = xyz + (size_t)b * 3 * NN;
  float px[4], py[4], pz[4], dist[4];
  int idx[4];
  for (int t = 0; t < 4; ++t) {
    int n = t * 1024 + tid;
    idx[t] = n;
    px[t] = xb[n]; py[t] = xb[NN + n]; pz[t] = xb[2 * NN + n];
    dist[t] = 1e10f;
  }
  __shared__ float bc[3];
  __shared__ float wd[32];
  __shared__ int   wi[32];
  if (tid == 0) { bc[0] = xb[0]; bc[1] = xb[NN]; bc[2] = xb[2 * NN]; }
  __syncthreads();
  for (int it = 0; it < SS; ++it) {
    float cx = bc[0], cy = bc[1], cz = bc[2];
    if (tid == 0) {
      float* q = query + ((size_t)b * SS + it) * 3;
      q[0] = cx; q[1] = cy; q[2] = cz;
      outq[(size_t)b * 3 * SS + it]          = cx;   // query^T (B,3,S)
      outq[(size_t)b * 3 * SS + SS + it]     = cy;
      outq[(size_t)b * 3 * SS + 2 * SS + it] = cz;
    }
    float bd = -1.f; int bi = 0x7fffffff;
    for (int t = 0; t < 4; ++t) {
      float dx = px[t] - cx, dy = py[t] - cy, dz = pz[t] - cz;
      float d = dx * dx + dy * dy + dz * dz;
      dist[t] = fminf(dist[t], d);
      if (dist[t] > bd || (dist[t] == bd && idx[t] < bi)) { bd = dist[t]; bi = idx[t]; }
    }
    for (int off = 16; off > 0; off >>= 1) {           // wave32 shuffle argmax
      float od = __shfl_xor(bd, off, 32);
      int   oi = __shfl_xor(bi, off, 32);
      if (od > bd || (od == bd && oi < bi)) { bd = od; bi = oi; }
    }
    int wid = tid >> 5;
    if ((tid & 31) == 0) { wd[wid] = bd; wi[wid] = bi; }
    __syncthreads();
    if (tid < 32) {                                    // wave-0 second-level argmax
      float gd = wd[tid]; int gi = wi[tid];
      for (int off = 16; off > 0; off >>= 1) {
        float od = __shfl_xor(gd, off, 32);
        int   oi = __shfl_xor(gi, off, 32);
        if (od > gd || (od == gd && oi < gi)) { gd = od; gi = oi; }
      }
      if (tid == 0) { bc[0] = xb[gi]; bc[1] = xb[NN + gi]; bc[2] = xb[2 * NN + gi]; }
    }
    __syncthreads();
  }
}

// ---- async global->LDS staging helper (gfx1250 GV mode, tracked by ASYNCcnt) ----
__device__ __forceinline__ void async_ld_b32(uint32_t lds_off, const float* gptr) {
  asm volatile("global_load_async_to_lds_b32 %0, %1, off"
               :: "v"(lds_off), "v"(gptr) : "memory");
}
__device__ __forceinline__ void wait_async0() {
  asm volatile("s_wait_asynccnt 0x0" ::: "memory");
}

// -------- ball query fused with grouping: forward scan == sort(where)[:k] --------
__global__ __launch_bounds__(256) void ball_group_kernel(const float* __restrict__ xyz,
                                                         const float* __restrict__ pts,
                                                         const float* __restrict__ query,
                                                         _Float16* __restrict__ X0,
                                                         int K, float r2) {
  int b = blockIdx.y, tid = threadIdx.x;
  int s = blockIdx.x * 256 + tid;
  const float* xb = xyz + (size_t)b * 3 * NN;
  const float* pb = pts + (size_t)b * 3 * NN;
  const float* q  = query + ((size_t)b * SS + s) * 3;
  float qx = q[0], qy = q[1], qz = q[2];
  __shared__ __align__(16) float sx[256], sy[256], sz[256], p0[256], p1[256], p2[256];
  size_t Msp = (size_t)BB * K * SS;
  int cnt = 0;
  float f6[6] = {0.f, 0.f, 0.f, 0.f, 0.f, 0.f};
  for (int n0 = 0; n0 < NN; n0 += 256) {
    __syncthreads();
    async_ld_b32((uint32_t)(uintptr_t)&sx[tid], xb + n0 + tid);
    async_ld_b32((uint32_t)(uintptr_t)&sy[tid], xb + NN + n0 + tid);
    async_ld_b32((uint32_t)(uintptr_t)&sz[tid], xb + 2 * NN + n0 + tid);
    async_ld_b32((uint32_t)(uintptr_t)&p0[tid], pb + n0 + tid);
    async_ld_b32((uint32_t)(uintptr_t)&p1[tid], pb + NN + n0 + tid);
    async_ld_b32((uint32_t)(uintptr_t)&p2[tid], pb + 2 * NN + n0 + tid);
    wait_async0();
    __syncthreads();
    for (int j = 0; j < 256; ++j) {
      if (cnt >= K) break;
      float dx = sx[j] - qx, dy = sy[j] - qy, dz = sz[j] - qz;
      float d2 = dx * dx + dy * dy + dz * dz;
      if (d2 <= r2) {
        size_t m = ((size_t)(b * K + cnt)) * SS + s;
        X0[0 * Msp + m] = (_Float16)p0[j];
        X0[1 * Msp + m] = (_Float16)p1[j];
        X0[2 * Msp + m] = (_Float16)p2[j];
        X0[3 * Msp + m] = (_Float16)dx;
        X0[4 * Msp + m] = (_Float16)dy;
        X0[5 * Msp + m] = (_Float16)dz;
        if (cnt == 0) { f6[0]=p0[j]; f6[1]=p1[j]; f6[2]=p2[j]; f6[3]=dx; f6[4]=dy; f6[5]=dz; }
        ++cnt;
      }
    }
  }
  for (int kk = cnt; kk < K; ++kk) {                   // pad with first hit
    size_t m = ((size_t)(b * K + kk)) * SS + s;
    for (int c = 0; c < 6; ++c) X0[c * Msp + m] = (_Float16)f6[c];
  }
}

// ------- WMMA GEMM: D[COUT x M] = W[COUT x CIN] * X[CIN x M], full-Cout blocks -------
// Block: 64 M-columns x COUT; 8 waves = 2(m-group) x 4(n-group); B frag reused across
// all Cout strips -> X read exactly once from HBM; WMMAs fully unrolled.
template<int CIN, int COUT>
__global__ __launch_bounds__(256) void conv_wmma_kernel(const _Float16* __restrict__ X,
                                                        const float* __restrict__ W,
                                                        _Float16* __restrict__ Y,
                                                        int MspI) {
  constexpr int NK = (CIN + 31) / 32;
  constexpr int NM = COUT / 32;   // strips per wave: strip = 2*i + wm
  size_t Msp = (size_t)MspI;
  __shared__ __align__(16) _Float16 As[COUT * 40];   // 80B rows: 16B-aligned frags
  __shared__ __align__(16) _Float16 Xs[64 * 40];
  int tid = threadIdx.x, lane = tid & 31, wid = tid >> 5;
  int wm = wid >> 2, wn = wid & 3;
  size_t n0 = (size_t)blockIdx.x * 64;
  v8f acc[NM];
#pragma unroll
  for (int i = 0; i < NM; ++i) acc[i] = (v8f){};
#pragma unroll
  for (int kc = 0; kc < NK; ++kc) {
    int kb = kc * 32;
    for (int i = tid; i < COUT * 32; i += 256) {       // stage W tile, f32->f16, pad K
      int m = i >> 5, k = i & 31, gk = kb + k;
      float w = (gk < CIN) ? W[m * CIN + gk] : 0.f;
      As[m * 40 + k] = (_Float16)w;
    }
    for (int i = tid; i < 32 * 64; i += 256) {         // stage X tile transposed [N][K]
      int k = i >> 6, n = i & 63, gk = kb + k;
      _Float16 v = (_Float16)0.f;
      if (gk < CIN) v = X[(size_t)gk * Msp + n0 + n];
      Xs[n * 40 + k] = v;
    }
    __syncthreads();
    // B fragment (32x16): lane holds column, K contiguous per half-wave
    int col = (lane & 15) + wn * 16;
    int kr = (lane < 16) ? 0 : 16;
    v8h blo = *(const v8h*)&Xs[col * 40 + kr];
    v8h bhi = *(const v8h*)&Xs[col * 40 + kr + 8];
    v16h bm;
#pragma unroll
    for (int j = 0; j < 8; ++j) { bm[j] = blo[j]; bm[j + 8] = bhi[j]; }
    int kbase = (lane < 16) ? 0 : 16;
#pragma unroll
    for (int i = 0; i < NM; ++i) {
      int row = (lane & 15) + (2 * i + wm) * 16;       // A frag per Cout strip
      v8h alo = *(const v8h*)&As[row * 40 + kbase];
      v8h ahi = *(const v8h*)&As[row * 40 + kbase + 8];
      v16h a;
#pragma unroll
      for (int j = 0; j < 8; ++j) { a[j] = alo[j]; a[j + 8] = ahi[j]; }
      acc[i] = __builtin_amdgcn_wmma_f32_16x16x32_f16(false, a, false, bm,
                                                      (short)0, acc[i], false, false);
    }
    __syncthreads();
  }
  size_t ng = n0 + (size_t)(wn * 16 + (lane & 15));
#pragma unroll
  for (int i = 0; i < NM; ++i) {
    int mg0 = (2 * i + wm) * 16 + ((lane < 16) ? 0 : 8);
#pragma unroll
    for (int j = 0; j < 8; ++j)
      Y[(size_t)(mg0 + j) * Msp + ng] = (_Float16)acc[i][j];
  }
}

// ---------------- BN stats: one block per channel, deterministic reduction ----------------
__global__ __launch_bounds__(256) void bn_stats_kernel(const _Float16* __restrict__ Y,
                                                       const float* __restrict__ gamma,
                                                       const float* __restrict__ beta,
                                                       float* __restrict__ ss, int MspI) {
  int c = blockIdx.x, tid = threadIdx.x;
  size_t Msp = (size_t)MspI;
  const _Float16* yc = Y + (size_t)c * Msp;
  float s1 = 0.f, s2 = 0.f;
  for (size_t i = tid; i < Msp; i += 256) { float v = (float)yc[i]; s1 += v; s2 += v * v; }
  __shared__ float r1[256], r2[256];
  r1[tid] = s1; r2[tid] = s2;
  __syncthreads();
  for (int off = 128; off > 0; off >>= 1) {
    if (tid < off) { r1[tid] += r1[tid + off]; r2[tid] += r2[tid + off]; }
    __syncthreads();
  }
  if (tid == 0) {
    float mean = r1[0] / (float)Msp;
    float var  = r2[0] / (float)Msp - mean * mean;
    float sc   = gamma[c] * rsqrtf(var + 1e-5f);
    ss[2 * c]     = sc;
    ss[2 * c + 1] = beta[c] - mean * sc;
  }
}

// ---------------- fused BN + ReLU, in-place, 8-wide f16 ----------------
__global__ __launch_bounds__(256) void bn_apply_kernel(_Float16* __restrict__ Y,
                                                       const float* __restrict__ ss,
                                                       int Cout, int MspI) {
  size_t Msp = (size_t)MspI;
  size_t tot8 = (size_t)Cout * Msp / 8;
  size_t stride = (size_t)gridDim.x * 256;
  for (size_t i = (size_t)blockIdx.x * 256 + threadIdx.x; i < tot8; i += stride) {
    size_t e = i * 8;
    int c = (int)(e / Msp);
    float sc = ss[2 * c], sh = ss[2 * c + 1];
    v8h y = *(v8h*)(Y + e);
    for (int j = 0; j < 8; ++j) {
      float f = fmaxf(0.f, (float)y[j] * sc + sh);
      y[j] = (_Float16)f;
    }
    *(v8h*)(Y + e) = y;
  }
}

// ---------------- max over k, write f32 slice of new_points ----------------
__global__ __launch_bounds__(256) void maxpool_kernel(const _Float16* __restrict__ X,
                                                      float* __restrict__ out,
                                                      int K, int Cout, int cbase) {
  int gid = blockIdx.x * 256 + threadIdx.x;
  int tot = BB * Cout * SS;
  if (gid >= tot) return;
  int s = gid & (SS - 1);
  int c = (gid >> 10) % Cout;
  int b = gid / (Cout * SS);
  size_t Msp = (size_t)BB * K * SS;
  const _Float16* xc = X + (size_t)c * Msp + (size_t)b * K * SS + s;
  float best = -1e30f;
  for (int kk = 0; kk < K; ++kk) best = fmaxf(best, (float)xc[(size_t)kk * SS]);
  out[((size_t)b * 320 + cbase + c) * SS + s] = best;
}

static void launch_conv(int Cin, int Cout, const _Float16* X, const float* W,
                        _Float16* Y, int MspI, hipStream_t st) {
  dim3 g(MspI / 64);
  if      (Cin ==  6 && Cout ==  32) conv_wmma_kernel< 6, 32><<<g,256,0,st>>>(X,W,Y,MspI);
  else if (Cin == 32 && Cout ==  32) conv_wmma_kernel<32, 32><<<g,256,0,st>>>(X,W,Y,MspI);
  else if (Cin == 32 && Cout ==  64) conv_wmma_kernel<32, 64><<<g,256,0,st>>>(X,W,Y,MspI);
  else if (Cin ==  6 && Cout ==  64) conv_wmma_kernel< 6, 64><<<g,256,0,st>>>(X,W,Y,MspI);
  else if (Cin == 64 && Cout ==  64) conv_wmma_kernel<64, 64><<<g,256,0,st>>>(X,W,Y,MspI);
  else if (Cin == 64 && Cout ==  96) conv_wmma_kernel<64, 96><<<g,256,0,st>>>(X,W,Y,MspI);
  else if (Cin == 64 && Cout == 128) conv_wmma_kernel<64,128><<<g,256,0,st>>>(X,W,Y,MspI);
  else if (Cin == 96 && Cout == 128) conv_wmma_kernel<96,128><<<g,256,0,st>>>(X,W,Y,MspI);
}

extern "C" void kernel_launch(void* const* d_in, const int* in_sizes, int n_in,
                              void* d_out, int out_size, void* d_ws, size_t ws_size,
                              hipStream_t stream) {
  (void)in_sizes; (void)n_in; (void)out_size; (void)ws_size;
  const float* xyz = (const float*)d_in[0];   // (B,3,N)
  const float* pts = (const float*)d_in[1];   // (B,3,N)
  float* out = (float*)d_out;

  char* ws = (char*)d_ws;
  size_t off = 0;
  float* query  = (float*)(ws + off); off += (size_t)BB * SS * 3 * sizeof(float);
  float* sstats = (float*)(ws + off); off += 256 * sizeof(float);
  off = (off + 255) & ~(size_t)255;
  _Float16* ping = (_Float16*)(ws + off); off += (size_t)96  * 524288 * 2;  // <=96 ch
  off = (off + 255) & ~(size_t)255;
  _Float16* pong = (_Float16*)(ws + off);                                   // <=128 ch

  fps_kernel<<<BB, 1024, 0, stream>>>(xyz, query, out);

  const int mlp[3][4] = {{6, 32, 32, 64}, {6, 64, 64, 128}, {6, 64, 96, 128}};
  const float rad[3]  = {0.2f, 0.4f, 0.8f};
  const int   ks[3]   = {16, 32, 128};
  int cbase = 0;
  for (int sc = 0; sc < 3; ++sc) {
    int K = ks[sc];
    int MspI = BB * K * SS;
    float r2 = rad[sc] * rad[sc];
    ball_group_kernel<<<dim3(SS / 256, BB), 256, 0, stream>>>(xyz, pts, query, ping, K, r2);
    _Float16* cur = ping; _Float16* nxt = pong;
    for (int l = 0; l < 3; ++l) {
      int Cin = mlp[sc][l], Cout = mlp[sc][l + 1];
      const float* Wp = (const float*)d_in[2 + (sc * 3 + l) * 4 + 0];
      const float* gp = (const float*)d_in[2 + (sc * 3 + l) * 4 + 2];
      const float* bp = (const float*)d_in[2 + (sc * 3 + l) * 4 + 3];
      launch_conv(Cin, Cout, cur, Wp, nxt, MspI, stream);
      bn_stats_kernel<<<Cout, 256, 0, stream>>>(nxt, gp, bp, sstats, MspI);
      size_t tot8 = (size_t)Cout * MspI / 8;
      int g = (int)((tot8 + 255) / 256); if (g > 8192) g = 8192;
      bn_apply_kernel<<<g, 256, 0, stream>>>(nxt, sstats, Cout, MspI);
      _Float16* t = cur; cur = nxt; nxt = t;
    }
    int CoutL = mlp[sc][3];
    maxpool_kernel<<<(BB * CoutL * SS) / 256, 256, 0, stream>>>(
        cur, out + (size_t)BB * 3 * SS, K, CoutL, cbase);
    cbase += CoutL;
  }
}
